// ImagePatchNAM_20126216750046
// MI455X (gfx1250) — matrix operand order repaired and checked by reading, not compiled, not run on hardware
//
#include <hip/hip_runtime.h>

typedef __attribute__((ext_vector_type(16))) _Float16 v16h;
typedef __attribute__((ext_vector_type(8)))  float    v8f;

#define B_    1024
#define C_    3
#define H_    224
#define W_    220
#define ROWS_ 4
#define COLS_ 5
#define P_    20
#define HID_  32
#define PH_   56   // H_/ROWS_
#define PW_   44   // W_/COLS_

// ---------------------------------------------------------------------------
// Phase 1: patch mean pooling (memory-bound: 605 MB read @ 23.3 TB/s ~ 26us).
// grid = B*ROWS (4096 blocks), block = 220 threads.
// Thread t owns fixed float4 column segment seg=t%55 (11 segs per patch col,
// PW_=44=11*4 so each float4 is entirely inside one patch column) and strides
// the 168 (c,row) lines with ro=t/55. One scalar accumulator per thread,
// then LDS reduction -> 5 patch means per block.
// ---------------------------------------------------------------------------
__global__ __launch_bounds__(220) void nam_pool_kernel(
    const float* __restrict__ x, float* __restrict__ feats) {
  __shared__ float lds[220];
  const int blk = blockIdx.x;
  const int b   = blk >> 2;          // / ROWS_
  const int r   = blk & 3;           // % ROWS_
  const int t   = threadIdx.x;
  const int seg = t % 55;            // float4 segment within 220-wide row
  const int ro  = t / 55;            // row offset 0..3

  const float* base = x + (size_t)b * (C_ * H_ * W_)
                        + (size_t)(r * PH_) * W_
                        + seg * 4;

  float acc = 0.f;
#pragma unroll 4
  for (int rowIdx = ro; rowIdx < C_ * PH_; rowIdx += 4) {
    const int c  = rowIdx / PH_;
    const int hr = rowIdx - c * PH_;
    const float4 v = *(const float4*)(base + ((size_t)c * H_ + hr) * W_);
    acc += v.x + v.y + v.z + v.w;
  }
  lds[t] = acc;
  __syncthreads();

  if (t < COLS_) {            // t = patch column index
    float s = 0.f;
#pragma unroll
    for (int ro2 = 0; ro2 < 4; ++ro2)
#pragma unroll
      for (int j = 0; j < 11; ++j)
        s += lds[ro2 * 55 + t * 11 + j];
    feats[b * P_ + r * COLS_ + t] = s * (1.0f / (float)(C_ * PH_ * PW_));
  }
}

// ---------------------------------------------------------------------------
// Phase 2: per-patch scalar MLP via V_WMMA_F32_16X16X32_F16 (K = HID = 32).
// grid = B/16 (64 blocks), block = 32 (one wave).
// A[16b x 32h] = relu(f*w1+b1) computed in f32, stored f16.
// B[32h x 16n] = w2[p,:] broadcast across all 16 columns -> every D column
// holds the 16 contribs for this (btile, p). Accumulate in f32.
// A f16 layout (ISA 7.12.2): lane<16 -> M=lane, K in {0..7}U{16..23};
//                            lane>=16 -> M=lane-16, K in {8..15}U{24..31}.
// B f16 layout: lane<16 -> K in {0..7}U{16..23} (2/VGPR), lane>=16 -> K in {8..15}U{24..31}.
// D f32 layout: lane 0 -> (N=0, M=0..7 in d[0..7]); lane 16 -> (N=0, M=8..15).
// ---------------------------------------------------------------------------
__global__ __launch_bounds__(32) void nam_mlp_wmma_kernel(
    const float* __restrict__ feats,
    const float* __restrict__ w1, const float* __restrict__ b1,
    const float* __restrict__ w2, const float* __restrict__ b2,
    float* __restrict__ out) {
  const int lane  = threadIdx.x;
  const int row16 = lane & 15;
  const int hi    = lane >> 4;
  const int btile = blockIdx.x * 16;
  const int b     = btile + row16;

  float sum_b2 = 0.f;
#pragma unroll
  for (int p = 0; p < P_; ++p) sum_b2 += b2[p];

  v8f score = {};

  for (int p = 0; p < P_; ++p) {
    const float  f   = feats[b * P_ + p];
    const float* w1p = w1 + p * HID_;
    const float* b1p = b1 + p * HID_;
    const float* w2p = w2 + p * HID_;

    // A matrix (activations, f16)
    v16h a;
    const int k0 = hi ? 8 : 0;
#pragma unroll
    for (int j = 0; j < 8; ++j) {
      const int kp = (j < 4) ? (k0 + 2 * j) : (16 + k0 + 2 * (j - 4));
      float a0 = f * w1p[kp]     + b1p[kp];
      float a1 = f * w1p[kp + 1] + b1p[kp + 1];
      a[2 * j]     = (_Float16)(a0 > 0.f ? a0 : 0.f);
      a[2 * j + 1] = (_Float16)(a1 > 0.f ? a1 : 0.f);
    }

    // B matrix (w2 broadcast across N, f16)
    v16h bm;
    const int kb0 = hi ? 8 : 0;
#pragma unroll
    for (int j = 0; j < 8; ++j) {
      const int kq = (j < 4) ? (kb0 + 2 * j) : (16 + kb0 + 2 * (j - 4));
      bm[2 * j]     = (_Float16)w2p[kq];
      bm[2 * j + 1] = (_Float16)w2p[kq + 1];
    }

    v8f cz = {};
    v8f d = __builtin_amdgcn_wmma_f32_16x16x32_f16(
        /*neg_a=*/false, a, /*neg_b=*/false, bm,
        /*c_mod=*/(short)0, cz, /*reuse_a=*/false, /*reuse_b=*/false);

    score += d;

    if (row16 == 0) {                 // lanes 0 and 16: column N=0 of D
      const float bb2 = b2[p];
#pragma unroll
      for (int i = 0; i < 8; ++i) {
        const int bo = btile + hi * 8 + i;
        out[B_ + bo * P_ + p] = d[i] + bb2;   // contribs after score block
      }
    }
  }

  if (row16 == 0) {
#pragma unroll
    for (int i = 0; i < 8; ++i) {
      const int bo = btile + hi * 8 + i;
      out[bo] = score[i] + sum_b2;            // score[B]
    }
  }
}

extern "C" void kernel_launch(void* const* d_in, const int* in_sizes, int n_in,
                              void* d_out, int out_size, void* d_ws, size_t ws_size,
                              hipStream_t stream) {
  const float* x  = (const float*)d_in[0];
  const float* w1 = (const float*)d_in[1];
  const float* b1 = (const float*)d_in[2];
  const float* w2 = (const float*)d_in[3];
  const float* b2 = (const float*)d_in[4];
  float* out   = (float*)d_out;
  float* feats = (float*)d_ws;               // B_*P_ f32 = 80 KB scratch

  nam_pool_kernel<<<B_ * ROWS_, 220, 0, stream>>>(x, feats);
  nam_mlp_wmma_kernel<<<B_ / 16, 32, 0, stream>>>(feats, w1, b1, w2, b2, out);
}
